// FALCON_11733850653239
// MI455X (gfx1250) — compile-verified, matrix operand order checked by reading, not run on hardware
//
#include <hip/hip_runtime.h>
#include <hip/hip_bf16.h>

typedef __bf16 v16bf __attribute__((ext_vector_type(16)));
typedef float  v8f   __attribute__((ext_vector_type(8)));
typedef int    v2i   __attribute__((ext_vector_type(2)));

#define EMB  128
#define ETOT 1024
#define NE   1000

#if defined(__HIP_DEVICE_COMPILE__) && \
    __has_builtin(__builtin_amdgcn_global_load_async_to_lds_b64) && \
    __has_builtin(__builtin_amdgcn_s_wait_asynccnt)
#define HAVE_ASYNC_LDS 1
// builtin expects pointers to the 64-bit payload type (int2), addrspace-qualified
typedef __attribute__((address_space(1))) v2i g_v2i;
typedef __attribute__((address_space(3))) v2i l_v2i;
#else
#define HAVE_ASYNC_LDS 0
#endif

// ---------------------------------------------------------------------------
// Kernel A: WMMA bf16 GEMMs (f32 accumulate), K=128 as 4 fully-unrolled
// v_wmma_f32_16x16x32_bf16 steps. One wave per 16x16 output tile.
//   MODE 0: L[m][n] = sum_k (e_all[m,k]+r_emb[k]) * W0[n,k]      + b0[n]
//   MODE 1: R[m][n] = sum_k  e_all[m,k]           * W0[n,128+k]
// A-fragment lane layout (ISA 7.12.2, 16-bit A 16x32): lane&15 = M row,
// lane>=16 shifts K by +8; elements {0..7} = K kb..kb+7, {8..15} = kb+16..+23.
// ---------------------------------------------------------------------------
template <int MODE>
__global__ __launch_bounds__(32) void falcon_gemm_wmma(
    const float* __restrict__ e_table, const float* __restrict__ anon,
    const float* __restrict__ r_emb,  const float* __restrict__ W0,
    const float* __restrict__ b0, float* __restrict__ out)
{
    const int lane  = threadIdx.x;            // 0..31
    const int mtile = blockIdx.x * 16;        // 64 tiles over 1024 rows
    const int ntile = blockIdx.y * 16;        // 8 tiles over 128 cols
    const int mrow  = mtile + (lane & 15);
    const int ncol  = ntile + (lane & 15);
    const int koff  = (lane >> 4) * 8;

    const float* erow = (mrow < NE) ? (e_table + mrow * EMB)
                                    : (anon + (mrow - NE) * EMB);
    const float* wrow = W0 + ncol * (2 * EMB) + (MODE ? EMB : 0);

    v8f acc = {0.f, 0.f, 0.f, 0.f, 0.f, 0.f, 0.f, 0.f};

#pragma unroll
    for (int ks = 0; ks < 4; ++ks) {
        const int kb = ks * 32 + koff;        // 16B-aligned (koff in {0,8})

        float4 a0 = *(const float4*)(erow + kb);
        float4 a1 = *(const float4*)(erow + kb + 4);
        float4 a2 = *(const float4*)(erow + kb + 16);
        float4 a3 = *(const float4*)(erow + kb + 20);
        if (MODE == 0) {
            const float4 r0 = *(const float4*)(r_emb + kb);
            const float4 r1 = *(const float4*)(r_emb + kb + 4);
            const float4 r2 = *(const float4*)(r_emb + kb + 16);
            const float4 r3 = *(const float4*)(r_emb + kb + 20);
            a0.x += r0.x; a0.y += r0.y; a0.z += r0.z; a0.w += r0.w;
            a1.x += r1.x; a1.y += r1.y; a1.z += r1.z; a1.w += r1.w;
            a2.x += r2.x; a2.y += r2.y; a2.z += r2.z; a2.w += r2.w;
            a3.x += r3.x; a3.y += r3.y; a3.z += r3.z; a3.w += r3.w;
        }
        const float4 q0 = *(const float4*)(wrow + kb);
        const float4 q1 = *(const float4*)(wrow + kb + 4);
        const float4 q2 = *(const float4*)(wrow + kb + 16);
        const float4 q3 = *(const float4*)(wrow + kb + 20);

        v16bf a, b;
        a[0]=(__bf16)a0.x; a[1]=(__bf16)a0.y; a[2]=(__bf16)a0.z; a[3]=(__bf16)a0.w;
        a[4]=(__bf16)a1.x; a[5]=(__bf16)a1.y; a[6]=(__bf16)a1.z; a[7]=(__bf16)a1.w;
        a[8]=(__bf16)a2.x; a[9]=(__bf16)a2.y; a[10]=(__bf16)a2.z; a[11]=(__bf16)a2.w;
        a[12]=(__bf16)a3.x; a[13]=(__bf16)a3.y; a[14]=(__bf16)a3.z; a[15]=(__bf16)a3.w;
        b[0]=(__bf16)q0.x; b[1]=(__bf16)q0.y; b[2]=(__bf16)q0.z; b[3]=(__bf16)q0.w;
        b[4]=(__bf16)q1.x; b[5]=(__bf16)q1.y; b[6]=(__bf16)q1.z; b[7]=(__bf16)q1.w;
        b[8]=(__bf16)q2.x; b[9]=(__bf16)q2.y; b[10]=(__bf16)q2.z; b[11]=(__bf16)q2.w;
        b[12]=(__bf16)q3.x; b[13]=(__bf16)q3.y; b[14]=(__bf16)q3.z; b[15]=(__bf16)q3.w;

        acc = __builtin_amdgcn_wmma_f32_16x16x32_bf16(
                  false, a, false, b, (short)0, acc, false, false);
    }

#pragma unroll
    for (int r = 0; r < 8; ++r) {
        const int row = mtile + ((lane < 16) ? r : (8 + r));
        const int col = ntile + (lane & 15);
        float v = acc[r];
        if (MODE == 0) v += b0[col];
        out[row * EMB + col] = v;
    }
}

// ---------------------------------------------------------------------------
// Kernel B: c_fs[j] = sigmoid( sum_k W1[k]*leaky(cwlb[k] + R[j,k]) + b1 )
// cwlb = c_emb@Wl.T + b0 recomputed per block (trivial). Also zero-inits
// d_out so kernel C's uint atomicMax is correct on every graph replay.
// ---------------------------------------------------------------------------
__global__ __launch_bounds__(256) void falcon_cfs(
    const float* __restrict__ c_emb, const float* __restrict__ W0,
    const float* __restrict__ b0,    const float* __restrict__ W1,
    const float* __restrict__ b1,    const float* __restrict__ Rm,
    float* __restrict__ cfs, float* __restrict__ out_init)
{
    __shared__ float cwlb[EMB];
    __shared__ float w1s[EMB];
    const int t = threadIdx.x;

    if (t < EMB) {
        float s = 0.f;
#pragma unroll 4
        for (int k = 0; k < EMB; ++k)
            s = fmaf(c_emb[k], W0[t * (2 * EMB) + k], s);
        cwlb[t] = s + b0[t];
        w1s[t]  = W1[t];
    }
    __syncthreads();

    const int j = blockIdx.x * 256 + t;
    const float* rrow = Rm + j * EMB;
    float s = 0.f;
#pragma unroll 4
    for (int k = 0; k < EMB; ++k) {
        const float x  = cwlb[k] + rrow[k];
        const float lk = fmaxf(x, 0.1f * x);
        s = fmaf(w1s[k], lk, s);
    }
    cfs[j]      = 1.0f / (1.0f + __expf(-(s + b1[0])));
    out_init[j] = 0.0f;
}

// ---------------------------------------------------------------------------
// Kernel C: pairwise core (runtime bottleneck, ~540 MFLOP VALU).
//   out[i] = max_j sigmoid(sum_k W1[k]*leaky(L[i,k]+R[j,k]) + b1) * cfs[j]
// R-chunk (256 rows x 128) staged in LDS via async-to-LDS b64 copies
// (ASYNCcnt path); row pad 130 dwords -> bank = (2j + k) mod 64, conflict-free
// and 8B-aligned for the b64 async writes. One i per wave, lanes sweep j.
// ---------------------------------------------------------------------------
#define JT   256
#define RPAD 130

__global__ __launch_bounds__(256) void falcon_pair(
    const float* __restrict__ L, const float* __restrict__ Rm,
    const float* __restrict__ W1, const float* __restrict__ b1,
    const float* __restrict__ cfs, float* __restrict__ out)
{
    __shared__ float sR[JT * RPAD];   // 130 KB
    __shared__ float sL[8 * EMB];     // 4 KB
    __shared__ float sW[EMB];
    __shared__ float sC[JT];

    const int t  = threadIdx.x;
    const int i0 = blockIdx.x * 8;
    const int j0 = blockIdx.y * JT;

#if HAVE_ASYNC_LDS
    // 256 rows x 64 float2 copies, 256 threads -> 64 async b64 ops per thread
    for (int idx = t; idx < JT * (EMB / 2); idx += 256) {
        const int jj = idx >> 6;              // row within chunk
        const int kk = (idx & 63) << 1;       // even k offset
        const float* g = Rm + (j0 + jj) * EMB + kk;
        float* l = &sR[jj * RPAD + kk];
        __builtin_amdgcn_global_load_async_to_lds_b64(
            (g_v2i*)g, (l_v2i*)l, 0, 0);
    }
#else
    for (int idx = t; idx < JT * EMB; idx += 256) {
        const int jj = idx >> 7, kk = idx & (EMB - 1);
        sR[jj * RPAD + kk] = Rm[(j0 + jj) * EMB + kk];
    }
#endif
    for (int idx = t; idx < 8 * EMB; idx += 256)
        sL[idx] = L[i0 * EMB + idx];
    if (t < EMB) sW[t] = W1[t];
    if (t < JT)  sC[t] = cfs[j0 + t];
#if HAVE_ASYNC_LDS
    __builtin_amdgcn_s_wait_asynccnt(0);
#endif
    __syncthreads();

    const int wave = t >> 5, lane = t & 31;
    const float b1s = b1[0];
    const float* lrow = sL + wave * EMB;      // uniform-address broadcast reads
    float best = 0.0f;

    for (int jg = 0; jg < JT / 32; ++jg) {
        const int jl = jg * 32 + lane;
        const float* rr = sR + jl * RPAD;
        float acc = 0.f;
#pragma unroll 8
        for (int k = 0; k < EMB; ++k) {
            const float x  = lrow[k] + rr[k];
            const float lk = fmaxf(x, 0.1f * x);
            acc = fmaf(sW[k], lk, acc);
        }
        const float v = 1.0f / (1.0f + __expf(-(acc + b1s)));
        best = fmaxf(best, v * sC[jl]);
    }

#pragma unroll
    for (int off = 16; off > 0; off >>= 1)
        best = fmaxf(best, __shfl_xor(best, off, 32));

    if (lane == 0) {
        // all candidate values are positive -> float bits are order-preserving
        atomicMax((unsigned int*)&out[i0 + wave], __float_as_uint(best));
    }
}

// ---------------------------------------------------------------------------
extern "C" void kernel_launch(void* const* d_in, const int* in_sizes, int n_in,
                              void* d_out, int out_size, void* d_ws, size_t ws_size,
                              hipStream_t stream) {
    const float* anon = (const float*)d_in[0];   // (24,128)
    const float* etab = (const float*)d_in[1];   // (1000,128)
    const float* cemb = (const float*)d_in[2];   // (128,)
    const float* remb = (const float*)d_in[3];   // (128,)
    const float* W0   = (const float*)d_in[4];   // (128,256)
    const float* b0   = (const float*)d_in[5];   // (128,)
    const float* W1   = (const float*)d_in[6];   // (128,)
    const float* b1   = (const float*)d_in[7];   // (1,)
    float* out = (float*)d_out;                  // (1024,)

    float* L   = (float*)d_ws;                   // 1024*128 f32
    float* R   = L + ETOT * EMB;                 // 1024*128 f32
    float* cfs = R + ETOT * EMB;                 // 1024    f32

    dim3 gA(ETOT / 16, EMB / 16);                // (64, 8)
    falcon_gemm_wmma<0><<<gA, 32, 0, stream>>>(etab, anon, remb, W0, b0, L);
    falcon_gemm_wmma<1><<<gA, 32, 0, stream>>>(etab, anon, remb, W0, b0, R);

    falcon_cfs<<<ETOT / 256, 256, 0, stream>>>(cemb, W0, b0, W1, b1, R, cfs, out);

    dim3 gC(ETOT / 8, ETOT / JT);                // (128, 4)
    falcon_pair<<<gC, 256, 0, stream>>>(L, R, W1, b1, cfs, out);
}
